// Classifier_57492432224697
// MI455X (gfx1250) — compile-verified, matrix operand order checked by reading, not run on hardware
//
#include <hip/hip_runtime.h>
#include <math.h>

// ---------------- types ----------------
typedef __bf16 bf16;
typedef bf16  v16bf __attribute__((ext_vector_type(16)));
typedef bf16  v8bf  __attribute__((ext_vector_type(8)));
typedef bf16  v4bf  __attribute__((ext_vector_type(4)));
typedef float v8f   __attribute__((ext_vector_type(8)));
typedef float v4f   __attribute__((ext_vector_type(4)));

#define WMMA_BF16(a, b, c) \
  __builtin_amdgcn_wmma_f32_16x16x32_bf16(false, (a), false, (b), (short)0, (c), false, false)

__device__ inline bf16 cvt_bf16(float f) { return (bf16)f; }  // native v_cvt path

// Branchless GELU (tanh/sigmoid form), uses hardware exp2 + rcp (TRANS ops,
// co-execute with WMMA). gelu(x) = x * sigmoid(1.5957691x + 0.0713548x^3)
__device__ inline float gelu_fast(float x) {
  float x2 = x * x;
  float g  = x * (0.7978845608028654f + 0.035677408136300125f * x2);
  float e  = __builtin_amdgcn_exp2f(g * -2.8853900817779268f);  // exp(-2g)
  return x * __builtin_amdgcn_rcpf(1.0f + e);
}

// Build a 16x32 bf16 A-fragment from a bf16 row-major tile in LDS.
// lane<16: K pattern {0..7,16..23}; lane>=16: {8..15,24..31}; m = lane&15.
__device__ inline v16bf make_afrag_bf16(const bf16* base, int stride, int kt, int lane) {
  int m  = lane & 15;
  int c0 = (lane < 16) ? 0 : 8;
  const bf16* row = base + m * stride + kt * 32 + c0;
  v8bf h0 = *(const v8bf*)(row);        // ds_load_b128
  v8bf h1 = *(const v8bf*)(row + 16);   // ds_load_b128
  v16bf a;
#pragma unroll
  for (int i = 0; i < 8; ++i) { a[i] = h0[i]; a[8 + i] = h1[i]; }
  return a;
}

// Pre-packed B fragment: 512 bf16 per (kt,nt) fragment, 16 contiguous per lane.
__device__ inline v16bf load_bfrag(const bf16* p, int frag, int lane) {
  return *(const v16bf*)(p + (size_t)frag * 512 + lane * 16);
}

// ---------------- kernel 1: pack fp32 weights into bf16 WMMA B-fragment layout ----------------
__global__ void pack_bfrag(const float* __restrict__ src, bf16* __restrict__ dst, int K, int N) {
  int p = blockIdx.x * blockDim.x + threadIdx.x;
  if (p >= K * N) return;
  int frag = p >> 9;
  int r    = p & 511;
  int l    = r >> 4;
  int j    = r & 15;
  int ntiles = N >> 4;
  int nt = frag % ntiles;
  int kt = frag / ntiles;
  int kloc = ((j < 8) ? j : j + 8) + ((l < 16) ? 0 : 8);
  int k = kt * 32 + kloc;
  int n = nt * 16 + (l & 15);
  dst[p] = cvt_bf16(src[(size_t)k * N + n]);
}

// ---------------- kernel 2: persistent 100-step ODE (one 64-row tile per WG, per net) ----------------
// LDS: z fp32 [64][516] master, z bf16 [64][520] A-feed, h bf16 [64][264].
__global__ void __launch_bounds__(256) ode_kernel(
    const float* __restrict__ z0,
    const float* __restrict__ W1r, const float* __restrict__ b1r, const float* __restrict__ b2r,
    const float* __restrict__ W1f, const float* __restrict__ b1f, const float* __restrict__ b2f,
    const bf16* __restrict__ pW1r, const bf16* __restrict__ pW2r,
    const bf16* __restrict__ pW1f, const bf16* __restrict__ pW2f,
    float* __restrict__ combined) {
  extern __shared__ char smem[];
  const int ZS = 516, ZBS = 520, HS = 264;
  float* zsh = (float*)smem;                               // 64*516*4 = 132096 B
  bf16*  zbf = (bf16*)(smem + 64 * ZS * 4);                // 64*520*2 =  66560 B
  bf16*  hsh = (bf16*)(smem + 64 * ZS * 4 + 64 * ZBS * 2); // 64*264*2 =  33792 B

  const int net = blockIdx.y;
  const float* W1 = net ? W1f : W1r;   // only row 512 (time column) read directly
  const float* b1 = net ? b1f : b1r;
  const float* b2 = net ? b2f : b2r;
  const bf16* pW1 = net ? pW1f : pW1r;
  const bf16* pW2 = net ? pW2f : pW2r;

  const int r0   = blockIdx.x * 64;
  const int tid  = threadIdx.x;
  const int lane = tid & 31;
  const int w    = tid >> 5;

  // cooperative load of the z tile; fill fp32 master + bf16 A-feed copy
  for (int i = tid; i < 64 * 128; i += 256) {
    int m = i >> 7;
    int k = (i & 127) << 2;
    v4f v = *(const v4f*)(z0 + (size_t)(r0 + m) * 512 + k);
    *(v4f*)(zsh + m * ZS + k) = v;
    v4bf b;
#pragma unroll
    for (int q = 0; q < 4; ++q) b[q] = cvt_bf16(v[q]);
    *(v4bf*)(zbf + m * ZBS + k) = b;
  }
  __syncthreads();

  // per-lane bias preloads for this wave's N columns
  float b1v[2], w1tv[2];
#pragma unroll
  for (int j = 0; j < 2; ++j) {
    int n = (2 * w + j) * 16 + (lane & 15);
    b1v[j]  = b1[n];
    w1tv[j] = W1[512 * 256 + n];   // time column of W1
  }
  float b2v[4];
#pragma unroll
  for (int j = 0; j < 4; ++j) {
    int n = (4 * w + j) * 16 + (lane & 15);
    b2v[j] = b2[n];
  }

  const float dt = 0.01f;
  const v8f zero8 = {0.f, 0.f, 0.f, 0.f, 0.f, 0.f, 0.f, 0.f};

  for (int step = 0; step < 100; ++step) {
    float tcoef = 1.0f - (float)step * dt;  // time input = 1 - i/100

    // ---- GEMM1: h = gelu(z @ W1a + b1 + tcoef*w1t)   (M=64, N=256, K=512)
    v8f acc1[4][2];
#pragma unroll
    for (int mt = 0; mt < 4; ++mt)
#pragma unroll
      for (int j = 0; j < 2; ++j) acc1[mt][j] = zero8;

    for (int kt = 0; kt < 16; ++kt) {
      v16bf a[4];
#pragma unroll
      for (int mt = 0; mt < 4; ++mt) a[mt] = make_afrag_bf16(zbf + mt * 16 * ZBS, ZBS, kt, lane);
      v16bf bb[2];
#pragma unroll
      for (int j = 0; j < 2; ++j) bb[j] = load_bfrag(pW1, kt * 16 + (2 * w + j), lane);
#pragma unroll
      for (int mt = 0; mt < 4; ++mt)
#pragma unroll
        for (int j = 0; j < 2; ++j) acc1[mt][j] = WMMA_BF16(a[mt], bb[j], acc1[mt][j]);
    }

#pragma unroll
    for (int mt = 0; mt < 4; ++mt) {
#pragma unroll
      for (int j = 0; j < 2; ++j) {
        float bias = b1v[j] + tcoef * w1tv[j];
        int n     = (2 * w + j) * 16 + (lane & 15);
        int mbase = mt * 16 + ((lane >> 4) << 3);
#pragma unroll
        for (int i = 0; i < 8; ++i) {
          float hv = gelu_fast(acc1[mt][j][i] + bias);
          hsh[(mbase + i) * HS + n] = cvt_bf16(hv);
        }
      }
    }
    __syncthreads();

    // ---- GEMM2: z -= dt * (h @ W2 + b2)   (M=64, N=512, K=256)
    v8f acc2[4][4];
#pragma unroll
    for (int mt = 0; mt < 4; ++mt)
#pragma unroll
      for (int j = 0; j < 4; ++j) acc2[mt][j] = zero8;

    for (int kt = 0; kt < 8; ++kt) {
      v16bf a[4];
#pragma unroll
      for (int mt = 0; mt < 4; ++mt) a[mt] = make_afrag_bf16(hsh + mt * 16 * HS, HS, kt, lane);
      v16bf bb[4];
#pragma unroll
      for (int j = 0; j < 4; ++j) bb[j] = load_bfrag(pW2, kt * 32 + (4 * w + j), lane);
#pragma unroll
      for (int mt = 0; mt < 4; ++mt)
#pragma unroll
        for (int j = 0; j < 4; ++j) acc2[mt][j] = WMMA_BF16(a[mt], bb[j], acc2[mt][j]);
    }

    // z update: fp32 master RMW + refresh bf16 A-feed copy (each element owned by one wave)
#pragma unroll
    for (int mt = 0; mt < 4; ++mt) {
#pragma unroll
      for (int j = 0; j < 4; ++j) {
        int n     = (4 * w + j) * 16 + (lane & 15);
        int mbase = mt * 16 + ((lane >> 4) << 3);
#pragma unroll
        for (int i = 0; i < 8; ++i) {
          int m = mbase + i;
          float zv = zsh[m * ZS + n];
          zv = zv - dt * (acc2[mt][j][i] + b2v[j]);
          zsh[m * ZS + n] = zv;
          zbf[m * ZBS + n] = cvt_bf16(zv);
        }
      }
    }
    __syncthreads();
  }

  // write tile into combined [B, 2*512] fp32 in workspace
  for (int i = tid; i < 64 * 128; i += 256) {
    int m = i >> 7;
    int k = (i & 127) << 2;
    v4f v = *(const v4f*)(zsh + m * ZS + k);
    *(v4f*)(combined + (size_t)(r0 + m) * 1024 + net * 512 + k) = v;
  }
}

// ---------------- kernel 3: classifier MLP + fused 2-column head ----------------
// LDS: A tile bf16 [64][1032] (converted once, shared by all waves) + logits[128].
__global__ void __launch_bounds__(256) mlp_kernel(
    const float* __restrict__ combined,   // [8192][1024] fp32
    const bf16*  __restrict__ pW1,        // mlp_W1 packed B-fragments (K=1024, N=1024)
    const float* __restrict__ b1,         // [1024]
    const float* __restrict__ W2,         // [1024][2]
    const float* __restrict__ b2,         // [2]
    float* __restrict__ out) {            // [8192][2]
  extern __shared__ char smem[];
  const int AS = 1032;
  bf16*  abf  = (bf16*)smem;                      // 64*1032*2 = 132096 B
  float* slog = (float*)(smem + 64 * AS * 2);     // 128 floats

  const int r0   = blockIdx.x * 64;
  const int tid  = threadIdx.x;
  const int lane = tid & 31;
  const int w    = tid >> 5;

  // cooperative fp32 -> bf16 conversion of the A tile (each element once)
  for (int i = tid; i < 64 * 256; i += 256) {
    int m = i >> 8;
    int k = (i & 255) << 2;
    v4f v = *(const v4f*)(combined + (size_t)(r0 + m) * 1024 + k);
    v4bf b;
#pragma unroll
    for (int q = 0; q < 4; ++q) b[q] = cvt_bf16(v[q]);
    *(v4bf*)(abf + m * AS + k) = b;
  }
  if (tid < 128) slog[tid] = 0.0f;
  __syncthreads();

  float lp[4][8][2];
#pragma unroll
  for (int mt = 0; mt < 4; ++mt)
#pragma unroll
    for (int i = 0; i < 8; ++i) { lp[mt][i][0] = 0.f; lp[mt][i][1] = 0.f; }

  const v8f zero8 = {0.f, 0.f, 0.f, 0.f, 0.f, 0.f, 0.f, 0.f};

  for (int j2 = 0; j2 < 4; ++j2) {
    v8f acc[4][2];
#pragma unroll
    for (int mt = 0; mt < 4; ++mt)
#pragma unroll
      for (int j = 0; j < 2; ++j) acc[mt][j] = zero8;

    for (int kt = 0; kt < 32; ++kt) {
      v16bf a[4];
#pragma unroll
      for (int mt = 0; mt < 4; ++mt)
        a[mt] = make_afrag_bf16(abf + mt * 16 * AS, AS, kt, lane);
      v16bf bb[2];
#pragma unroll
      for (int j = 0; j < 2; ++j)
        bb[j] = load_bfrag(pW1, kt * 64 + (w * 8 + 2 * j2 + j), lane);
#pragma unroll
      for (int mt = 0; mt < 4; ++mt)
#pragma unroll
        for (int j = 0; j < 2; ++j) acc[mt][j] = WMMA_BF16(a[mt], bb[j], acc[mt][j]);
    }

#pragma unroll
    for (int mt = 0; mt < 4; ++mt) {
#pragma unroll
      for (int j = 0; j < 2; ++j) {
        int n = (w * 8 + 2 * j2 + j) * 16 + (lane & 15);
        float bias = b1[n];
        float w20 = W2[n * 2 + 0];
        float w21 = W2[n * 2 + 1];
#pragma unroll
        for (int i = 0; i < 8; ++i) {
          float hv = gelu_fast(acc[mt][j][i] + bias);
          lp[mt][i][0] += hv * w20;
          lp[mt][i][1] += hv * w21;
        }
      }
    }
  }

  // reduce the N dimension across the 16 lanes of each half-wave
#pragma unroll
  for (int mt = 0; mt < 4; ++mt)
#pragma unroll
    for (int i = 0; i < 8; ++i)
#pragma unroll
      for (int c = 0; c < 2; ++c) {
        float v = lp[mt][i][c];
        for (int mask = 1; mask < 16; mask <<= 1) v += __shfl_xor(v, mask, 32);
        lp[mt][i][c] = v;
      }

  if ((lane & 15) == 0) {
#pragma unroll
    for (int mt = 0; mt < 4; ++mt)
#pragma unroll
      for (int i = 0; i < 8; ++i)
#pragma unroll
        for (int c = 0; c < 2; ++c) {
          int m = mt * 16 + ((lane >> 4) << 3) + i;
          atomicAdd(&slog[m * 2 + c], lp[mt][i][c]);  // ds_add_f32 across waves
        }
  }
  __syncthreads();
  if (tid < 128) {
    int m = tid >> 1, c = tid & 1;
    out[(size_t)(r0 + m) * 2 + c] = slog[tid] + b2[c];
  }
}

// ---------------- host launcher ----------------
extern "C" void kernel_launch(void* const* d_in, const int* in_sizes, int n_in,
                              void* d_out, int out_size, void* d_ws, size_t ws_size,
                              hipStream_t stream) {
  const float* z       = (const float*)d_in[0];
  const float* real_W1 = (const float*)d_in[1];
  const float* real_b1 = (const float*)d_in[2];
  const float* real_W2 = (const float*)d_in[3];
  const float* real_b2 = (const float*)d_in[4];
  const float* fake_W1 = (const float*)d_in[5];
  const float* fake_b1 = (const float*)d_in[6];
  const float* fake_W2 = (const float*)d_in[7];
  const float* fake_b2 = (const float*)d_in[8];
  const float* mlp_W1  = (const float*)d_in[9];
  const float* mlp_b1  = (const float*)d_in[10];
  const float* mlp_W2  = (const float*)d_in[11];
  const float* mlp_b2  = (const float*)d_in[12];
  float* out = (float*)d_out;

  char* ws = (char*)d_ws;
  bf16* pW1r = (bf16*)(ws + 0);          // 512x256 bf16 = 256 KB
  bf16* pW1f = (bf16*)(ws + 262144);
  bf16* pW2r = (bf16*)(ws + 524288);     // 256x512
  bf16* pW2f = (bf16*)(ws + 786432);
  bf16* pM1  = (bf16*)(ws + 1048576);    // 1024x1024 = 2 MB
  float* combined = (float*)(ws + (size_t)(4u << 20));  // 8192x1024 fp32 = 32 MB

  // one-time weight packing (bf16, WMMA B-fragment layout)
  pack_bfrag<<<512, 256, 0, stream>>>(real_W1, pW1r, 512, 256);  // rows 0..511 only
  pack_bfrag<<<512, 256, 0, stream>>>(fake_W1, pW1f, 512, 256);
  pack_bfrag<<<512, 256, 0, stream>>>(real_W2, pW2r, 256, 512);
  pack_bfrag<<<512, 256, 0, stream>>>(fake_W2, pW2f, 256, 512);
  pack_bfrag<<<4096, 256, 0, stream>>>(mlp_W1, pM1, 1024, 1024);

  // persistent ODE: 128 tiles x 2 nets, z tile lives in LDS for all 100 steps
  size_t lds_ode = (size_t)64 * 516 * 4 + (size_t)64 * 520 * 2 + (size_t)64 * 264 * 2; // 232448 B
  ode_kernel<<<dim3(128, 2), 256, lds_ode, stream>>>(
      z, real_W1, real_b1, real_b2, fake_W1, fake_b1, fake_b2,
      pW1r, pW2r, pW1f, pW2f, combined);

  // classifier
  size_t lds_mlp = (size_t)64 * 1032 * 2 + 128 * 4;  // 132608 B
  mlp_kernel<<<128, 256, lds_mlp, stream>>>(combined, pM1, mlp_b1, mlp_W2, mlp_b2, out);
}
